// AttentionLayerY_13331578486949
// MI455X (gfx1250) — matrix operand order repaired and checked by reading, not compiled, not run on hardware
//
#include <hip/hip_runtime.h>
#include <hip/hip_bf16.h>

// Problem dims
#define B_  32
#define T_  2048
#define E_  1024
#define D_  1024
#define H_  1024

typedef __attribute__((ext_vector_type(16))) __bf16 v16bf;
typedef __attribute__((ext_vector_type(8)))  float  v8f;
typedef __attribute__((ext_vector_type(8)))  unsigned short v8us;

union FragBF {
  v16bf bf;
  v8us  u[2];
};

__device__ __forceinline__ unsigned int f2bf(float f) {
  // round-to-nearest-even f32 -> bf16 (returned in low 16 bits)
  unsigned int u = __float_as_uint(f);
  return (u + 0x7FFFu + ((u >> 16) & 1u)) >> 16;
}

__device__ __forceinline__ float fast_tanh(float x) {
#if __has_builtin(__builtin_amdgcn_tanhf)
  return __builtin_amdgcn_tanhf(x);          // V_TANH_F32 (gfx1250 TRANS op)
#else
  // branchless: tanh(x) = 1 - 2/(e^{2x}+1); exact limits at +/-inf
  float e = __expf(2.0f * x);
#if __has_builtin(__builtin_amdgcn_rcpf)
  return 1.0f - 2.0f * __builtin_amdgcn_rcpf(e + 1.0f);
#else
  return 1.0f - 2.0f / (e + 1.0f);
#endif
#endif
}

// ---------------------------------------------------------------------------
// Kernel 0: W_enc [E][H] f32 -> Wt [H][E] bf16 (transposed, fragment friendly)
// ---------------------------------------------------------------------------
__global__ __launch_bounds__(256)
void wenc_transpose_kernel(const float* __restrict__ W, unsigned short* __restrict__ Wt) {
  __shared__ float tile[64][65];
  const int e0  = blockIdx.x * 64;
  const int h0  = blockIdx.y * 64;
  const int tid = threadIdx.x;
  const int r   = tid >> 4;          // 0..15
  const int c4  = (tid & 15) * 4;    // 0..60
#pragma unroll
  for (int i = 0; i < 4; ++i) {
    const int rr = r + i * 16;
    float4 f = *(const float4*)(W + (size_t)(e0 + rr) * H_ + h0 + c4);
    tile[rr][c4 + 0] = f.x; tile[rr][c4 + 1] = f.y;
    tile[rr][c4 + 2] = f.z; tile[rr][c4 + 3] = f.w;
  }
  __syncthreads();
#pragma unroll
  for (int i = 0; i < 4; ++i) {
    const int rr = r + i * 16;       // local h
    unsigned int lo = f2bf(tile[c4 + 0][rr]) | (f2bf(tile[c4 + 1][rr]) << 16);
    unsigned int hi = f2bf(tile[c4 + 2][rr]) | (f2bf(tile[c4 + 3][rr]) << 16);
    uint2 pk; pk.x = lo; pk.y = hi;
    *(uint2*)(Wt + (size_t)(h0 + rr) * E_ + e0 + c4) = pk;
  }
}

// ---------------------------------------------------------------------------
// Kernel 1: hidden_dec[b][h] = dec[b] . W_dec[:,h] + b_dec[h]   (f32, tiny)
// ---------------------------------------------------------------------------
__global__ __launch_bounds__(256)
void dec_hidden_kernel(const float* __restrict__ dec, const float* __restrict__ W_dec,
                       const float* __restrict__ b_dec, float* __restrict__ hdec) {
  const int h = blockIdx.x * 256 + threadIdx.x;
  const int b = blockIdx.y;
  const float* dr = dec + (size_t)b * D_;
  float acc = b_dec[h];
#pragma unroll 4
  for (int k = 0; k < D_; ++k)
    acc = fmaf(dr[k], W_dec[(size_t)k * H_ + h], acc);
  hdec[(size_t)b * H_ + h] = acc;
}

// ---------------------------------------------------------------------------
// Kernel 2: fused  logits[b,t] = tanh(enc[b,t,:]@W_enc + b_enc + hdec[b,:]) . w_out
//           + b_out, then mask.  bf16 WMMA, f32 accumulate, [B,T,H] never hits
//           memory.  128 t-rows per WG (8 waves x 16 rows), LDS-staged tiles,
//           next-tile latency hidden via global_prefetch (CDNA5).
// ---------------------------------------------------------------------------
#define T_TILE   128
#define A_STRIDE 1032   // 1024 + 8 bf16 pad (breaks bank conflicts)
#define B_STRIDE 1032
#define NTILES   (H_ / 16)

__global__ __launch_bounds__(256)
void fused_score_kernel(const float* __restrict__ enc,
                        const unsigned short* __restrict__ Wt,   // [H][E] bf16
                        const float* __restrict__ b_enc,
                        const float* __restrict__ hdec,          // [B][H]
                        const float* __restrict__ w_out,
                        const float* __restrict__ b_out,
                        const float* __restrict__ inp_mask,
                        float* __restrict__ logits) {
  extern __shared__ unsigned short smem[];
  unsigned short* As = smem;                        // [T_TILE][A_STRIDE] bf16
  unsigned short* Bt = smem + T_TILE * A_STRIDE;    // [16][B_STRIDE]     bf16
  float* hbS = (float*)(smem + T_TILE * A_STRIDE + 16 * B_STRIDE); // [H] b_enc+hdec
  float* wS  = hbS + H_;                                            // [H] w_out

  const int tid = threadIdx.x;
  const int b   = blockIdx.y;
  const int t0  = blockIdx.x * T_TILE;

  // prefetch B tile 0 (one 128B line per thread = 32KB tile)
  __builtin_prefetch((const char*)Wt + (size_t)tid * 128, 0, 3);

  // ---- stage A: enc[b, t0..t0+127, :] -> bf16 LDS; stage hb/w vectors ----
  {
    const float* encB = enc + (size_t)(b * T_ + t0) * E_;
    for (int idx = tid * 4; idx < T_TILE * E_; idx += 256 * 4) {
      const int row = idx >> 10;
      const int col = idx & (E_ - 1);
      float4 f = *(const float4*)(encB + (size_t)row * E_ + col);
      uint2 p;
      p.x = f2bf(f.x) | (f2bf(f.y) << 16);
      p.y = f2bf(f.z) | (f2bf(f.w) << 16);
      *(uint2*)(As + row * A_STRIDE + col) = p;
    }
    for (int i = tid; i < H_; i += 256) {
      hbS[i] = b_enc[i] + hdec[(size_t)b * H_ + i];
      wS[i]  = w_out[i];
    }
  }
  __syncthreads();               // A/hb/w staged

  const int lane = tid & 31;
  const int wave = tid >> 5;          // 8 waves, 16 t-rows each
  const int hi16 = lane >> 4;         // lane half
  const int lid  = lane & 15;

  // A 16-bit 16x32 layout: lanes 0-15 K{0..7,16..23}, lanes 16-31 K{8..15,24..31}
  const unsigned short* Arow = As + (wave * 16 + lid) * A_STRIDE + hi16 * 8;
  // B layout: lane = column n, lanes 0-15 K{0..7,16..23}, lanes 16-31 K{8..15,24..31}
  const unsigned short* Brow = Bt + lid * B_STRIDE + hi16 * 8;

  float part[8];
#pragma unroll
  for (int v = 0; v < 8; ++v) part[v] = 0.f;

  for (int nt = 0; nt < NTILES; ++nt) {
    const int n0 = nt * 16;

    // stage B-tile: Wt[n0..n0+15][0..E) bf16 -> LDS (prefetched -> cache hits)
    {
      const unsigned short* src = Wt + (size_t)n0 * E_;
#pragma unroll
      for (int j = 0; j < 8; ++j) {
        const int i = tid + j * 256;
        const int n = i >> 7;        // row within 16-wide tile
        const int c = i & 127;       // 16B chunk within row
        *(uint4*)(Bt + n * B_STRIDE + c * 8) =
            *(const uint4*)(src + (size_t)n * E_ + c * 8);
      }
    }
    __syncthreads();

    // prefetch the NEXT tile's lines; latency hidden under 32 WMMAs
    if (nt + 1 < NTILES)
      __builtin_prefetch((const char*)(Wt + (size_t)(n0 + 16) * E_) + (size_t)tid * 128, 0, 3);

    v8f acc = {0.f, 0.f, 0.f, 0.f, 0.f, 0.f, 0.f, 0.f};
#pragma unroll 4
    for (int kk = 0; kk < E_; kk += 32) {
      FragBF a, bb;
      a.u[0]  = *(const v8us*)(Arow + kk);
      a.u[1]  = *(const v8us*)(Arow + kk + 16);
      bb.u[0] = *(const v8us*)(Brow + kk);
      bb.u[1] = *(const v8us*)(Brow + kk + 16);
      acc = __builtin_amdgcn_wmma_f32_16x16x32_bf16(false, a.bf, false, bb.bf,
                                                    (short)0, acc, false, false);
    }

    // C/D layout: VGPR v -> row M = v + hi16*8, col N = lid
    const float wv = wS[n0 + lid];
    const float hb = hbS[n0 + lid];
#pragma unroll
    for (int v = 0; v < 8; ++v)
      part[v] += fast_tanh(acc[v] + hb) * wv;

    __syncthreads();   // all waves done reading Bt before restaging
  }

  // reduce partial dot over the 16 lanes of each half, write masked logits
  const float bo = b_out[0];
#pragma unroll
  for (int v = 0; v < 8; ++v) {
    float s = part[v];
    s += __shfl_xor(s, 1, 32);
    s += __shfl_xor(s, 2, 32);
    s += __shfl_xor(s, 4, 32);
    s += __shfl_xor(s, 8, 32);
    if (lid == v) {
      const int t  = t0 + wave * 16 + hi16 * 8 + v;
      float lg = s + bo;
      lg = (inp_mask[(size_t)b * T_ + t] > 0.f) ? lg : -1e9f;
      logits[(size_t)b * T_ + t] = lg;
    }
  }
}

// ---------------------------------------------------------------------------
// Kernel 3: row softmax over logits [B][T] in place
// ---------------------------------------------------------------------------
__global__ __launch_bounds__(256)
void softmax_kernel(float* __restrict__ p) {
  __shared__ float red[256];
  const int b = blockIdx.x, tid = threadIdx.x;
  float* row = p + (size_t)b * T_;
  float v[8], m = -3.0e38f;
#pragma unroll
  for (int i = 0; i < 8; ++i) { v[i] = row[tid + i * 256]; m = fmaxf(m, v[i]); }
  red[tid] = m; __syncthreads();
  for (int s = 128; s > 0; s >>= 1) {
    if (tid < s) red[tid] = fmaxf(red[tid], red[tid + s]);
    __syncthreads();
  }
  m = red[0]; __syncthreads();
  float sum = 0.f;
#pragma unroll
  for (int i = 0; i < 8; ++i) { v[i] = __expf(v[i] - m); sum += v[i]; }
  red[tid] = sum; __syncthreads();
  for (int s = 128; s > 0; s >>= 1) {
    if (tid < s) red[tid] += red[tid + s];
    __syncthreads();
  }
  const float inv = 1.f / red[0];
#pragma unroll
  for (int i = 0; i < 8; ++i) row[tid + i * 256] = v[i] * inv;
}

// ---------------------------------------------------------------------------
// Kernel 4: attn[b,e] = sum_t probs[b,t] * enc[b,t,e]   (HBM-bound, coalesced)
// ---------------------------------------------------------------------------
__global__ __launch_bounds__(256)
void attn_kernel(const float* __restrict__ enc, const float* __restrict__ probs,
                 float* __restrict__ attn) {
  const int e = blockIdx.x * 256 + threadIdx.x;
  const int b = blockIdx.y;
  const float* encB = enc + (size_t)b * T_ * E_ + e;
  const float* pr   = probs + (size_t)b * T_;
  float acc = 0.f;
#pragma unroll 4
  for (int t = 0; t < T_; ++t)
    acc = fmaf(pr[t], encB[(size_t)t * E_], acc);
  attn[(size_t)b * E_ + e] = acc;
}

// ---------------------------------------------------------------------------
extern "C" void kernel_launch(void* const* d_in, const int* in_sizes, int n_in,
                              void* d_out, int out_size, void* d_ws, size_t ws_size,
                              hipStream_t stream) {
  (void)in_sizes; (void)n_in; (void)out_size; (void)ws_size;

  const float* enc   = (const float*)d_in[0];   // [B,T,E]
  const float* dec   = (const float*)d_in[1];   // [B,D]
  const float* mask  = (const float*)d_in[2];   // [B,T]
  const float* W_enc = (const float*)d_in[3];   // [E,H]
  const float* b_enc = (const float*)d_in[4];   // [H]
  const float* W_dec = (const float*)d_in[5];   // [D,H]
  const float* b_dec = (const float*)d_in[6];   // [H]
  const float* w_out = (const float*)d_in[7];   // [H]
  const float* b_out = (const float*)d_in[8];   // scalar

  float* out_attn  = (float*)d_out;             // [B,E]
  float* out_probs = out_attn + (size_t)B_ * E_;// [B,T] (logits -> probs in place)

  // workspace layout: Wt bf16 [H][E] (2 MB), then hdec f32 [B][H] (128 KB)
  unsigned short* Wt = (unsigned short*)d_ws;
  float* hdec = (float*)((char*)d_ws + (size_t)H_ * E_ * sizeof(unsigned short));

  wenc_transpose_kernel<<<dim3(E_ / 64, H_ / 64), 256, 0, stream>>>(W_enc, Wt);
  dec_hidden_kernel<<<dim3(H_ / 256, B_), 256, 0, stream>>>(dec, W_dec, b_dec, hdec);

  const size_t smem = (size_t)(T_TILE * A_STRIDE + 16 * B_STRIDE) * sizeof(unsigned short)
                    + (size_t)2 * H_ * sizeof(float);
  fused_score_kernel<<<dim3(T_ / T_TILE, B_), 256, smem, stream>>>(
      enc, Wt, b_enc, hdec, w_out, b_out, mask, out_probs);

  softmax_kernel<<<B_, 256, 0, stream>>>(out_probs);
  attn_kernel<<<dim3(E_ / 256, B_), 256, 0, stream>>>(enc, out_probs, out_attn);
}
